// H_DYNA_42348377538865
// MI455X (gfx1250) — matrix operand order repaired and checked by reading, not compiled, not run on hardware
//
#include <hip/hip_runtime.h>

// ---------------- problem dims ----------------
#define B_    32
#define T_    12
#define HOR_  12
#define N_    512
#define H_    64
#define P_    32
#define S_    12
#define ML_   64
#define MG_   32
#define DE_   10
#define BN_   (B_ * N_)      // 16384
#define KTOT_ (S_ * H_)      // 768

typedef _Float16 half_t;
typedef __attribute__((ext_vector_type(16))) _Float16 v16h;
typedef __attribute__((ext_vector_type(8)))  _Float16 v8h;
typedef __attribute__((ext_vector_type(8)))  float    v8f;

static __device__ inline v8f v8f_zero() {
    v8f z;
#pragma unroll
    for (int i = 0; i < 8; ++i) z[i] = 0.0f;
    return z;
}

// A-matrix fragment (f16, 16x32). ISA layout: lane L holds row M=L&15.
// lo lanes: halves 0..7 -> K=k0+0..7,   halves 8..15 -> K=k0+16..23
// hi lanes: halves 0..7 -> K=k0+8..15,  halves 8..15 -> K=k0+24..31
static __device__ inline v16h load_a_frag(const half_t* rowbase_k0, int hi) {
    v8h lo = *(const v8h*)(rowbase_k0 + 8 * hi);
    v8h hh = *(const v8h*)(rowbase_k0 + 16 + 8 * hi);
    v16h a;
#pragma unroll
    for (int i = 0; i < 8; ++i) { a[i] = lo[i]; a[i + 8] = hh[i]; }
    return a;
}

// B-matrix fragment (f16, 32x16). lane L holds column N=L&15;
// lo lanes: K=k0+0..7,16..23, hi lanes: K=k0+8..15,24..31 (same order as A).
static __device__ inline v16h load_b_frag(const half_t* colbase_k0, int hi) {
    const half_t* p = colbase_k0 + 8 * hi;
    v8h x0 = *(const v8h*)p;
    v8h x1 = *(const v8h*)(p + 16);
    v16h b;
#pragma unroll
    for (int i = 0; i < 8; ++i) { b[i] = x0[i]; b[i + 8] = x1[i]; }
    return b;
}

static __device__ inline v8f wmma16(v16h a, v16h b, v8f c) {
    return __builtin_amdgcn_wmma_f32_16x16x32_f16(false, a, false, b, (short)0, c,
                                                  false, false);
}

// ============================================================================
// Precompute kernels (run once per launch)
// ============================================================================
#define KLN_ (ML_ * KTOT_)           // 49152
#define KGN_ (MG_ * KTOT_)           // 24576
#define BIN_ 96
#define LMN_ (ML_ * P_)              // 2048
#define GMN_ (MG_ * P_)              // 1024
#define PREP_TOT_ (KLN_ + KGN_ + BIN_ + LMN_ + GMN_)

__global__ void k_prep_static(const float* __restrict__ lmem,
                              const float* __restrict__ gmem,
                              const float* __restrict__ Wq,
                              const float* __restrict__ bq,
                              half_t* __restrict__ Kloc, half_t* __restrict__ Kglob,
                              float* __restrict__ bias,
                              float* __restrict__ lmean, float* __restrict__ gmean) {
    int idx = blockIdx.x * blockDim.x + threadIdx.x;
    if (idx < KLN_) {
        int m = idx / KTOT_, k = idx % KTOT_, s = k / H_, hh = k % H_;
        float a = 0.f;
        for (int p = 0; p < P_; ++p) a += Wq[hh * P_ + p] * lmem[(m * S_ + s) * P_ + p];
        Kloc[idx] = (half_t)a;
    } else if (idx < KLN_ + KGN_) {
        int j = idx - KLN_;
        int m = j / KTOT_, k = j % KTOT_, s = k / H_, hh = k % H_;
        float a = 0.f;
        for (int p = 0; p < P_; ++p) a += Wq[hh * P_ + p] * gmem[(m * S_ + s) * P_ + p];
        Kglob[j] = (half_t)a;
    } else if (idx < KLN_ + KGN_ + BIN_) {
        int m = idx - KLN_ - KGN_;
        const float* mem = (m < ML_) ? (lmem + m * S_ * P_) : (gmem + (m - ML_) * S_ * P_);
        float a = 0.f;
        for (int sp = 0; sp < S_ * P_; ++sp) a += bq[sp % P_] * mem[sp];
        bias[m] = a;
    } else if (idx < KLN_ + KGN_ + BIN_ + LMN_) {
        int j = idx - KLN_ - KGN_ - BIN_;
        int m = j / P_, p = j % P_;
        float a = 0.f;
        for (int s = 0; s < S_; ++s) a += lmem[(m * S_ + s) * P_ + p];
        lmean[j] = a / (float)S_;
    } else if (idx < PREP_TOT_) {
        int j = idx - KLN_ - KGN_ - BIN_ - LMN_;
        int m = j / P_, p = j % P_;
        float a = 0.f;
        for (int s = 0; s < S_; ++s) a += gmem[(m * S_ + s) * P_ + p];
        gmean[j] = a / (float)S_;
    }
}

// nswT[n][h][f] = sum_d emb[n][d] * wp[d][f][h]   (f16, col-major for B frags)
__global__ void k_prep_nswT(const float* __restrict__ emb, const float* __restrict__ wp,
                            half_t* __restrict__ nswT) {
    int idx = blockIdx.x * blockDim.x + threadIdx.x;
    if (idx >= N_ * H_ * 64) return;
    int n = idx / (H_ * 64);
    int h = (idx / 64) % H_;
    int f = idx % 64;
    float a = 0.f;
    for (int d = 0; d < DE_; ++d) a += emb[n * DE_ + d] * wp[(d * 64 + f) * H_ + h];
    nswT[idx] = (half_t)a;
}

// WxT[o][k] = Wx[k+1][o] (drop the x-row; store col-major f16)
__global__ void k_prep_wT(const float* __restrict__ Wz, const float* __restrict__ Wr,
                          const float* __restrict__ Wc, half_t* __restrict__ WzT,
                          half_t* __restrict__ WrT, half_t* __restrict__ WcT) {
    int idx = blockIdx.x * blockDim.x + threadIdx.x;
    if (idx >= 3 * H_ * H_) return;
    int w = idx / (H_ * H_);
    int j = idx % (H_ * H_);
    int o = j / H_, k = j % H_;
    const float* W = (w == 0) ? Wz : (w == 1) ? Wr : Wc;
    half_t* WT = (w == 0) ? WzT : (w == 1) ? WrT : WcT;
    WT[j] = (half_t)W[(k + 1) * H_ + o];
}

// ============================================================================
// Per-step kernel 1: attention scores (WMMA GEMM [16 x 768]x[768 x 96])
//                    + softmax + context means -> fused[bn][64] (f16)
// ============================================================================
__global__ __launch_bounds__(32) void k_scores(
    const half_t* __restrict__ buf, const half_t* __restrict__ Kloc,
    const half_t* __restrict__ Kglob, const float* __restrict__ bias,
    const float* __restrict__ lmean, const float* __restrict__ gmean,
    half_t* __restrict__ fused, int step) {
    const int lane = threadIdx.x;
    const int hi = lane >> 4;
    const int lr = lane & 15;
    const int bn0 = blockIdx.x * 16;

    v8f acc[6];
#pragma unroll
    for (int t = 0; t < 6; ++t) acc[t] = v8f_zero();

    const half_t* Bbase[6];
#pragma unroll
    for (int t = 0; t < 6; ++t)
        Bbase[t] = (t < 4) ? (Kloc + (size_t)(t * 16 + lr) * KTOT_)
                           : (Kglob + (size_t)((t - 4) * 16 + lr) * KTOT_);

    for (int s = 0; s < S_; ++s) {
        int phys = (step + s) % S_;
        const half_t* arow = buf + ((size_t)phys * BN_ + bn0 + lr) * H_;
        // hide the 2MB circular-buffer slot hop for the next iteration
        if (s + 1 < S_) {
            int pn = (step + s + 1) % S_;
            __builtin_prefetch(buf + ((size_t)pn * BN_ + bn0 + lr) * H_, 0, 3);
        }
#pragma unroll
        for (int c = 0; c < 2; ++c) {
            int k0 = s * H_ + c * 32;
            // issue A + all 6 B-fragment loads before any WMMA so waits overlap
            v16h a = load_a_frag(arow + c * 32, hi);
            v16h bf[6];
#pragma unroll
            for (int t = 0; t < 6; ++t) bf[t] = load_b_frag(Bbase[t] + k0, hi);
#pragma unroll
            for (int t = 0; t < 6; ++t) acc[t] = wmma16(a, bf[t], acc[t]);
        }
    }

    __shared__ __align__(16) float sc[16][96];
#pragma unroll
    for (int t = 0; t < 6; ++t) {
        int col = t * 16 + lr;
        float bs = bias[col];
#pragma unroll
        for (int r = 0; r < 8; ++r) sc[r + 8 * hi][col] = acc[t][r] + bs;
    }
    __syncthreads();

    // softmax: lane (lr,hi) -> row lr, part hi (0: local/64, 1: global/32)
    {
        int row = lr;
        int off = hi ? 64 : 0;
        int cnt = hi ? MG_ : ML_;
        float mx = -1e30f;
        for (int m = 0; m < cnt; ++m) mx = fmaxf(mx, sc[row][off + m]);
        float sum = 0.f;
        for (int m = 0; m < cnt; ++m) {
            float e = __expf(sc[row][off + m] - mx);
            sc[row][off + m] = e;
            sum += e;
        }
        float inv = 1.f / sum;
        for (int m = 0; m < cnt; ++m) sc[row][off + m] *= inv;
    }
    __syncthreads();

    // fused = [attn_l @ lmean | attn_g @ gmean]  (f16 out)
    {
        int row = lr;
        if (!hi) {
            for (int p = 0; p < P_; ++p) {
                float a = 0.f;
                for (int m = 0; m < ML_; ++m) a += sc[row][m] * lmean[m * P_ + p];
                fused[(size_t)(bn0 + row) * H_ + p] = (half_t)a;
            }
        } else {
            for (int p = 0; p < P_; ++p) {
                float a = 0.f;
                for (int m = 0; m < MG_; ++m) a += sc[row][64 + m] * gmean[m * P_ + p];
                fused[(size_t)(bn0 + row) * H_ + P_ + p] = (half_t)a;
            }
        }
    }
}

// ============================================================================
// Per-step kernel 2: ctx[b][n][h] = fused[b][n][:] @ nsw[n]  (one wave per n)
// ============================================================================
__global__ __launch_bounds__(32) void k_ctx(const half_t* __restrict__ fused,
                                            const half_t* __restrict__ nswT,
                                            float* __restrict__ ctx) {
    const int lane = threadIdx.x;
    const int hi = lane >> 4;
    const int lr = lane & 15;
    const int n = blockIdx.x;

    v8f acc[2][4];
#pragma unroll
    for (int mt = 0; mt < 2; ++mt)
#pragma unroll
        for (int nt = 0; nt < 4; ++nt) acc[mt][nt] = v8f_zero();

    const half_t* nb = nswT + (size_t)n * H_ * 64;
#pragma unroll
    for (int c = 0; c < 2; ++c) {
        // batch every operand load of this k-chunk before the WMMA burst
        v16h a[2];
#pragma unroll
        for (int mt = 0; mt < 2; ++mt) {
            int b = mt * 16 + lr;
            a[mt] = load_a_frag(fused + ((size_t)b * N_ + n) * H_ + c * 32, hi);
        }
        v16h bf[4];
#pragma unroll
        for (int nt = 0; nt < 4; ++nt)
            bf[nt] = load_b_frag(nb + (size_t)(nt * 16 + lr) * 64 + c * 32, hi);
#pragma unroll
        for (int nt = 0; nt < 4; ++nt)
#pragma unroll
            for (int mt = 0; mt < 2; ++mt) acc[mt][nt] = wmma16(a[mt], bf[nt], acc[mt][nt]);
    }
#pragma unroll
    for (int mt = 0; mt < 2; ++mt)
#pragma unroll
        for (int nt = 0; nt < 4; ++nt) {
            int col = nt * 16 + lr;
#pragma unroll
            for (int r = 0; r < 8; ++r) {
                int b = mt * 16 + r + 8 * hi;
                ctx[((size_t)b * N_ + n) * H_ + col] = acc[mt][nt][r];
            }
        }
}

// ============================================================================
// Per-step kernel 3: GRU cell (WMMA for h@Wz/Wr and (r*h)@Wc), buffer update,
// decoder output y = h@Wo + bo (autoregressive input for next decode step).
// mode: 0 = encoder(x=src[:,step]), 1 = dec first(x=src[:,T-1]), 2 = dec(x=yprev)
// ============================================================================
__global__ __launch_bounds__(32) void k_gru(
    const float* __restrict__ src, const float* __restrict__ yprev,
    float* __restrict__ h, half_t* __restrict__ buf, const float* __restrict__ ctx,
    const half_t* __restrict__ WzT, const half_t* __restrict__ WrT,
    const half_t* __restrict__ WcT, const float* __restrict__ Wz,
    const float* __restrict__ bz, const float* __restrict__ Wr,
    const float* __restrict__ br, const float* __restrict__ Wc,
    const float* __restrict__ bc, const float* __restrict__ Wo,
    const float* __restrict__ bo, float* __restrict__ yout,
    float* __restrict__ dout, int step, int mode, int kdec) {
    const int lane = threadIdx.x;
    const int hi = lane >> 4;
    const int lr = lane & 15;
    const int bn0 = blockIdx.x * 16;

    __shared__ __align__(16) float h32[16 * 64];
    __shared__ __align__(16) half_t h16[16 * 64];
    __shared__ __align__(16) half_t rh16[16 * 64];
    __shared__ __align__(16) float hn[16 * 64];
    __shared__ float xv[16];

    for (int i = lane; i < 16 * 64; i += 32) {
        float v = h[(size_t)bn0 * H_ + i];
        h32[i] = v;
        h16[i] = (half_t)v;
    }
    if (lane < 16) {
        int bn = bn0 + lane, b = bn / N_, n = bn % N_;
        float x;
        if (mode == 0)      x = src[((size_t)b * T_ + step) * N_ + n];
        else if (mode == 1) x = src[((size_t)b * T_ + (T_ - 1)) * N_ + n];
        else                x = yprev[bn];
        xv[lane] = x;
    }
    __syncthreads();

    // z/r pre-activations: h @ Wz[1:], h @ Wr[1:]
    v8f zacc[4], racc[4];
#pragma unroll
    for (int nt = 0; nt < 4; ++nt) { zacc[nt] = v8f_zero(); racc[nt] = v8f_zero(); }
#pragma unroll
    for (int c = 0; c < 2; ++c) {
        v16h a = load_a_frag(&h16[lr * 64 + c * 32], hi);
        v16h bzf[4], brf[4];
#pragma unroll
        for (int nt = 0; nt < 4; ++nt) {
            int col = nt * 16 + lr;
            bzf[nt] = load_b_frag(WzT + (size_t)col * 64 + c * 32, hi);
            brf[nt] = load_b_frag(WrT + (size_t)col * 64 + c * 32, hi);
        }
#pragma unroll
        for (int nt = 0; nt < 4; ++nt) {
            zacc[nt] = wmma16(a, bzf[nt], zacc[nt]);
            racc[nt] = wmma16(a, brf[nt], racc[nt]);
        }
    }

    float zreg[4][8];
#pragma unroll
    for (int nt = 0; nt < 4; ++nt) {
        int col = nt * 16 + lr;
        float wz0 = Wz[col], wr0 = Wr[col];   // row 0 of [65,64] = x weights
        float bzv = bz[col], brv = br[col];
#pragma unroll
        for (int r = 0; r < 8; ++r) {
            int row = r + 8 * hi;
            float x = xv[row];
            float zp = zacc[nt][r] + x * wz0 + bzv;
            float rp = racc[nt][r] + x * wr0 + brv;
            float z = 1.f / (1.f + __expf(-zp));
            float rr = 1.f / (1.f + __expf(-rp));
            zreg[nt][r] = z;
            rh16[row * 64 + col] = (half_t)(rr * h32[row * 64 + col]);
        }
    }
    __syncthreads();

    // candidate pre-activation: (r*h) @ Wc[1:]
    v8f cacc[4];
#pragma unroll
    for (int nt = 0; nt < 4; ++nt) cacc[nt] = v8f_zero();
#pragma unroll
    for (int c = 0; c < 2; ++c) {
        v16h a = load_a_frag(&rh16[lr * 64 + c * 32], hi);
        v16h bcf[4];
#pragma unroll
        for (int nt = 0; nt < 4; ++nt)
            bcf[nt] = load_b_frag(WcT + (size_t)(nt * 16 + lr) * 64 + c * 32, hi);
#pragma unroll
        for (int nt = 0; nt < 4; ++nt) cacc[nt] = wmma16(a, bcf[nt], cacc[nt]);
    }

    const int w = step % S_;
#pragma unroll
    for (int nt = 0; nt < 4; ++nt) {
        int col = nt * 16 + lr;
        float wc0 = Wc[col], bcv = bc[col];
#pragma unroll
        for (int r = 0; r < 8; ++r) {
            int row = r + 8 * hi;
            float x = xv[row];
            float hp = h32[row * 64 + col];
            float hcp = cacc[nt][r] + x * wc0 + bcv + ctx[(size_t)(bn0 + row) * H_ + col];
            float hc = tanhf(hcp);
            float z = zreg[nt][r];
            float hv = (1.f - z) * hp + z * hc;
            h[(size_t)(bn0 + row) * H_ + col] = hv;
            buf[((size_t)w * BN_ + bn0 + row) * H_ + col] = (half_t)hv;
            hn[row * 64 + col] = hv;
        }
    }

    if (mode >= 1) {
        __syncthreads();
        if (lane < 16) {
            int bn = bn0 + lane, b = bn / N_, n = bn % N_;
            float a = bo[0];
            for (int hh = 0; hh < H_; ++hh) a += hn[lane * 64 + hh] * Wo[hh];
            yout[bn] = a;
            dout[((size_t)b * HOR_ + kdec) * N_ + n] = a;
        }
    }
}

// ============================================================================
// Host launcher
// ============================================================================
extern "C" void kernel_launch(void* const* d_in, const int* in_sizes, int n_in,
                              void* d_out, int out_size, void* d_ws, size_t ws_size,
                              hipStream_t stream) {
    (void)in_sizes; (void)n_in; (void)out_size; (void)ws_size;

    const float* src  = (const float*)d_in[0];
    const float* lmem = (const float*)d_in[1];
    const float* gmem = (const float*)d_in[2];
    const float* Wq   = (const float*)d_in[3];
    const float* bq   = (const float*)d_in[4];
    const float* emb  = (const float*)d_in[5];
    const float* wp   = (const float*)d_in[6];
    const float* Wz   = (const float*)d_in[7];
    const float* bz   = (const float*)d_in[8];
    const float* Wr   = (const float*)d_in[9];
    const float* br   = (const float*)d_in[10];
    const float* Wc   = (const float*)d_in[11];
    const float* bc   = (const float*)d_in[12];
    const float* Wo   = (const float*)d_in[13];
    const float* bo   = (const float*)d_in[14];
    float* out = (float*)d_out;

    char* ws = (char*)d_ws;
    size_t off = 0;
    auto take = [&](size_t bytes) {
        size_t o = off;
        off = (off + bytes + 255) & ~(size_t)255;
        return o;
    };
    half_t* buffer = (half_t*)(ws + take((size_t)S_ * BN_ * H_ * sizeof(half_t)));
    float*  hstate = (float*)(ws + take((size_t)BN_ * H_ * sizeof(float)));
    float*  ctx    = (float*)(ws + take((size_t)BN_ * H_ * sizeof(float)));
    half_t* fused  = (half_t*)(ws + take((size_t)BN_ * H_ * sizeof(half_t)));
    float*  ybuf   = (float*)(ws + take((size_t)BN_ * sizeof(float)));
    half_t* Kloc   = (half_t*)(ws + take((size_t)ML_ * KTOT_ * sizeof(half_t)));
    half_t* Kglob  = (half_t*)(ws + take((size_t)MG_ * KTOT_ * sizeof(half_t)));
    float*  bias   = (float*)(ws + take((size_t)96 * sizeof(float)));
    float*  lmean  = (float*)(ws + take((size_t)ML_ * P_ * sizeof(float)));
    float*  gmean  = (float*)(ws + take((size_t)MG_ * P_ * sizeof(float)));
    half_t* nswT   = (half_t*)(ws + take((size_t)N_ * H_ * 64 * sizeof(half_t)));
    half_t* WzT    = (half_t*)(ws + take((size_t)H_ * H_ * sizeof(half_t)));
    half_t* WrT    = (half_t*)(ws + take((size_t)H_ * H_ * sizeof(half_t)));
    half_t* WcT    = (half_t*)(ws + take((size_t)H_ * H_ * sizeof(half_t)));

    // zero-init recurrent state (graph-capturable memset nodes)
    hipMemsetAsync(buffer, 0, (size_t)S_ * BN_ * H_ * sizeof(half_t), stream);
    hipMemsetAsync(hstate, 0, (size_t)BN_ * H_ * sizeof(float), stream);

    // one-time precompute
    k_prep_static<<<(PREP_TOT_ + 255) / 256, 256, 0, stream>>>(
        lmem, gmem, Wq, bq, Kloc, Kglob, bias, lmean, gmean);
    k_prep_nswT<<<(N_ * H_ * 64 + 255) / 256, 256, 0, stream>>>(emb, wp, nswT);
    k_prep_wT<<<(3 * H_ * H_ + 255) / 256, 256, 0, stream>>>(Wz, Wr, Wc, WzT, WrT, WcT);

    // encoder
    for (int t = 0; t < T_; ++t) {
        k_scores<<<BN_ / 16, 32, 0, stream>>>(buffer, Kloc, Kglob, bias, lmean, gmean,
                                              fused, t);
        k_ctx<<<N_, 32, 0, stream>>>(fused, nswT, ctx);
        k_gru<<<BN_ / 16, 32, 0, stream>>>(src, ybuf, hstate, buffer, ctx, WzT, WrT, WcT,
                                           Wz, bz, Wr, br, Wc, bc, Wo, bo, ybuf, out, t,
                                           0, 0);
    }
    // decoder (autoregressive: x_{k+1} = y_k)
    for (int k = 0; k < HOR_; ++k) {
        int step = T_ + k;
        k_scores<<<BN_ / 16, 32, 0, stream>>>(buffer, Kloc, Kglob, bias, lmean, gmean,
                                              fused, step);
        k_ctx<<<N_, 32, 0, stream>>>(fused, nswT, ctx);
        k_gru<<<BN_ / 16, 32, 0, stream>>>(src, ybuf, hstate, buffer, ctx, WzT, WrT, WcT,
                                           Wz, bz, Wr, br, Wc, bc, Wo, bo, ybuf, out,
                                           step, (k == 0) ? 1 : 2, k);
    }
}
